// BlockAttnRes_18511309046434
// MI455X (gfx1250) — matrix operand order repaired and checked by reading, not compiled, not run on hardware
//
#include <hip/hip_runtime.h>
#include <math.h>

typedef __attribute__((ext_vector_type(16))) _Float16 v16h;
typedef __attribute__((ext_vector_type(8)))  _Float16 v8h;
typedef __attribute__((ext_vector_type(8)))  float    v8f;
typedef __attribute__((ext_vector_type(4)))  float    f32x4;

#define B_   4
#define N_   8
#define S_   4096
#define D_   1024
#define SD_  ((size_t)S_ * D_)
#define BSD_ ((size_t)B_ * SD_)
#define EPS_ 1.1920928955078125e-07f

// workspace layout (bytes)
#define WS_SUMM 0                           // float    [32][1024]  summaries
#define WS_A16  (WS_SUMM + 32 * 1024 * 4)   // _Float16 [16][1024]  padded A operand
#define WS_Q    (WS_A16  + 16 * 1024 * 2)   // float    [16][1024]  query (rows 0..3 valid)
#define WS_W    (WS_Q    + 16 * 1024 * 4)   // float    [32]        softmax weights

// ---------------------------------------------------------------------------
// Kernel 1: RMSNorm the 32 block-summary rows; build f16 A operand (16x1024,
// rows 4..15 zero) from partial_block[:, -1, :] + current[:, -1, :].
// ---------------------------------------------------------------------------
__global__ void __launch_bounds__(256)
prep_kernel(const float* __restrict__ current,
            const float* __restrict__ block_outputs,
            const float* __restrict__ partial_block,
            const float* __restrict__ rms_w,
            float* __restrict__ summ,        // [32][1024]
            _Float16* __restrict__ a16)      // [16][1024]
{
    const int blk = blockIdx.x;
    const int tid = threadIdx.x;             // 256 threads
    __shared__ float red[256];

    if (blk < 32) {                          // blk = b*8 + n
        const float* row = block_outputs + ((size_t)blk * S_ + (S_ - 1)) * D_;
        float x[4], ss = 0.0f;
        #pragma unroll
        for (int i = 0; i < 4; ++i) { x[i] = row[tid + 256 * i]; ss += x[i] * x[i]; }
        red[tid] = ss;
        __syncthreads();
        #pragma unroll
        for (int off = 128; off > 0; off >>= 1) {
            if (tid < off) red[tid] += red[tid + off];
            __syncthreads();
        }
        const float scale = rsqrtf(red[0] * (1.0f / D_) + EPS_);
        #pragma unroll
        for (int i = 0; i < 4; ++i) {
            const int d = tid + 256 * i;
            summ[(size_t)blk * D_ + d] = x[i] * scale * rms_w[d];
        }
    } else {                                 // blk 32..47 -> A16 row r
        const int r = blk - 32;
        #pragma unroll
        for (int i = 0; i < 4; ++i) {
            const int d = tid + 256 * i;
            float v = 0.0f;
            if (r < B_) {
                const size_t off = ((size_t)r * S_ + (S_ - 1)) * D_ + d;
                v = partial_block[off] + current[off];
            }
            a16[r * D_ + d] = (_Float16)v;
        }
    }
}

// ---------------------------------------------------------------------------
// Kernel 2: Q[16,1024] = A16[16,1024] x W^T via v_wmma_f32_16x16x32_f16.
// One wave per 16-column N-tile; K-loop of 32 steps of 32.
// ISA 7.12.2 16-bit layouts:
//   A (16x32): lane L: M=L&15, half=L>>4; elem j <-> K = 16*(j>>3) + 8*half + (j&7)
//   B (32x16): lane L: N=L&15, half=L>>4; elem j <-> K = 16*(j>>3) + 8*half + (j&7)
//   C/D f32  : vgpr r holds row M = r + 8*half, col N = L&15
// ---------------------------------------------------------------------------
__global__ void __launch_bounds__(32)
query_gemm_wmma(const _Float16* __restrict__ a16,  // [16][1024]
                const float* __restrict__ w,       // res_proj_w [1024][1024] row-major
                float* __restrict__ q)             // [16][1024]
{
    const int lane = threadIdx.x;
    const int half = lane >> 4;
    const int l15  = lane & 15;
    const int n0   = blockIdx.x * 16;

    v8f acc = {};
    const _Float16* arow = a16 + l15 * D_;                 // A row M = l15
    const float*    wrow = w + (size_t)(n0 + l15) * D_;    // b[k][n] = W[n][k]

    for (int k0 = 0; k0 < D_; k0 += 32) {
        // A operand: two contiguous 16B groups per lane
        v16h a;
        const v8h lo = *(const v8h*)(arow + k0 + 8 * half);
        const v8h hi = *(const v8h*)(arow + k0 + 16 + 8 * half);
        #pragma unroll
        for (int j = 0; j < 8; ++j) { a[j] = lo[j]; a[8 + j] = hi[j]; }

        // B operand: two groups of 8 K values of W row n (K = 8*half + j, 16 + 8*half + j), converted f32->f16
        v16h b;
        const float* wp = wrow + k0 + 8 * half;
        #pragma unroll
        for (int j = 0; j < 16; ++j) b[j] = (_Float16)wp[(j & 7) + 16 * (j >> 3)];

        acc = __builtin_amdgcn_wmma_f32_16x16x32_f16(
                  /*neg_a=*/false, a, /*neg_b=*/false, b,
                  /*c_mod=*/(short)0, acc, /*reuse_a=*/false, /*reuse_b=*/false);
    }

    #pragma unroll
    for (int r = 0; r < 8; ++r)
        q[(size_t)(r + 8 * half) * D_ + n0 + l15] = acc[r];
}

// ---------------------------------------------------------------------------
// Kernel 3: scores[b,n] = (Q[b] . summ[b*8+n]) / 32 ; softmax over n -> weights
// 32 waves, one (b,n) pair each; wave32 shuffle reduction.
// ---------------------------------------------------------------------------
__global__ void __launch_bounds__(1024)
scores_softmax(const float* __restrict__ q,     // [16][1024]
               const float* __restrict__ summ,  // [32][1024]
               float* __restrict__ wout)        // [32]
{
    __shared__ float s_sc[32];
    const int tid  = threadIdx.x;
    const int wv   = tid >> 5;          // 0..31 == b*8+n
    const int lane = tid & 31;
    const int b    = wv >> 3;

    const float* qr = q + (size_t)b * D_;
    const float* sr = summ + (size_t)wv * D_;
    float dot = 0.0f;
    for (int d = lane; d < D_; d += 32) dot += qr[d] * sr[d];
    #pragma unroll
    for (int off = 16; off > 0; off >>= 1) dot += __shfl_xor(dot, off, 32);
    if (lane == 0) s_sc[wv] = dot * (1.0f / 32.0f);   // 1/sqrt(1024)
    __syncthreads();

    if (tid < B_) {
        float s[8], m = -1e30f;
        #pragma unroll
        for (int n = 0; n < 8; ++n) { s[n] = s_sc[tid * 8 + n]; m = fmaxf(m, s[n]); }
        float sum = 0.0f;
        #pragma unroll
        for (int n = 0; n < 8; ++n) { s[n] = __expf(s[n] - m); sum += s[n]; }
        const float inv = 1.0f / sum;
        #pragma unroll
        for (int n = 0; n < 8; ++n) wout[tid * 8 + n] = s[n] * inv;
    }
}

// ---------------------------------------------------------------------------
// Kernel 4: the HBM-bound pass.
//   out0 = (partial_block + current) + sum_n w[b,n] * block_outputs[b,n]
//   out1 = (partial_block + current)
// grid.y = b so weights stay in registers; float4 NT loads/stores (streaming,
// working set 805 MB >> 192 MB L2).
// ---------------------------------------------------------------------------
__global__ void __launch_bounds__(256)
blend_kernel(const float* __restrict__ current,
             const float* __restrict__ block_outputs,
             const float* __restrict__ partial_block,
             const float* __restrict__ wts,     // [4][8]
             float* __restrict__ out)           // [2][B][S][D] concatenated
{
    const int b = blockIdx.y;
    float w[8];
    #pragma unroll
    for (int n = 0; n < 8; ++n) w[n] = wts[b * 8 + n];

    const size_t plane = (size_t)b * SD_;
    const f32x4* cur = (const f32x4*)(current + plane);
    const f32x4* pb  = (const f32x4*)(partial_block + plane);
    const f32x4* bo  = (const f32x4*)(block_outputs + (size_t)b * N_ * SD_);
    f32x4* o0 = (f32x4*)(out + plane);
    f32x4* o1 = (f32x4*)(out + BSD_ + plane);

    const size_t nvec    = SD_ / 4;     // float4s per [S,D] plane
    const size_t pstride = SD_ / 4;     // n-plane stride in float4s

    for (size_t i = (size_t)blockIdx.x * blockDim.x + threadIdx.x; i < nvec;
         i += (size_t)gridDim.x * blockDim.x) {
        f32x4 p = __builtin_nontemporal_load(cur + i) +
                  __builtin_nontemporal_load(pb  + i);
        f32x4 att = w[0] * __builtin_nontemporal_load(bo + i);
        #pragma unroll
        for (int n = 1; n < 8; ++n)
            att += w[n] * __builtin_nontemporal_load(bo + (size_t)n * pstride + i);
        __builtin_nontemporal_store(p + att, o0 + i);
        __builtin_nontemporal_store(p,       o1 + i);
    }
}

// ---------------------------------------------------------------------------
extern "C" void kernel_launch(void* const* d_in, const int* in_sizes, int n_in,
                              void* d_out, int out_size, void* d_ws, size_t ws_size,
                              hipStream_t stream) {
    (void)in_sizes; (void)n_in; (void)out_size; (void)ws_size;
    const float* current       = (const float*)d_in[0];
    const float* block_outputs = (const float*)d_in[1];
    const float* partial_block = (const float*)d_in[2];
    const float* res_proj_w    = (const float*)d_in[3];
    const float* rms_w         = (const float*)d_in[4];
    float* out = (float*)d_out;

    char* ws = (char*)d_ws;
    float*    summ = (float*)(ws + WS_SUMM);
    _Float16* a16  = (_Float16*)(ws + WS_A16);
    float*    q    = (float*)(ws + WS_Q);
    float*    wts  = (float*)(ws + WS_W);

    prep_kernel<<<48, 256, 0, stream>>>(current, block_outputs, partial_block,
                                        rms_w, summ, a16);
    query_gemm_wmma<<<64, 32, 0, stream>>>(a16, res_proj_w, q);
    scores_softmax<<<1, 1024, 0, stream>>>(q, summ, wts);
    dim3 grid(2048, 4);
    blend_kernel<<<grid, 256, 0, stream>>>(current, block_outputs, partial_block,
                                           wts, out);
}
